// Model_29686813950424
// MI455X (gfx1250) — matrix operand order repaired and checked by reading, not compiled, not run on hardware
//
#include <hip/hip_runtime.h>
#include <hip/hip_bf16.h>

// ---------------- problem constants (from reference) ----------------
#define B_   64
#define L_   1024
#define F_   900
#define E_   8
#define D_   768
#define SE_  64
#define EPS_ 1e-9f

// ---------------- GEMM tiling ----------------
#define TM 128
#define TN 128
#define TK 32
#define LDS_STRIDE 34   // TK + 2 bf16 pad

typedef __bf16 bf16_t;
typedef __attribute__((ext_vector_type(16))) __bf16 v16bf;
typedef __attribute__((ext_vector_type(8)))  float  v8f;

// =====================================================================
// Kernel 0: zero the guide-loss accumulator (ws is poisoned by harness)
// =====================================================================
__global__ void moe_zero(float* s_accum) {
    s_accum[0] = 0.0f;
}

// =====================================================================
// Kernel 1: per-batch softmax, mask, top-2 renorm, selection embedding,
//           guide-loss partial sum.  grid = B, block = SE (64 threads)
// =====================================================================
__global__ void moe_setup(const float* __restrict__ logits,   // [B,E]
                          const int*   __restrict__ masks,    // [B,E]
                          const float* __restrict__ selE,     // [B,E,SE]
                          float* __restrict__ s_accum,        // [1]
                          int*   __restrict__ eIdx,           // [B,2]
                          float* __restrict__ pVal,           // [B,2]
                          float* __restrict__ sel_out)        // [B,SE]
{
    const int b = blockIdx.x;
    const int t = threadIdx.x;          // 0..SE-1

    __shared__ float inact[E_];

    if (t == 0) {
        float lg[E_];
        float mx = -1e30f;
        for (int e = 0; e < E_; ++e) { lg[e] = logits[b * E_ + e]; mx = fmaxf(mx, lg[e]); }
        float sum = 0.f;
        for (int e = 0; e < E_; ++e) { lg[e] = __expf(lg[e] - mx); sum += lg[e]; }

        float p[E_], in_[E_];
        float smask = 0.f, inSum = 0.f;
        for (int e = 0; e < E_; ++e) {
            float raw = lg[e] / sum;
            float m   = (masks[b * E_ + e] == 1) ? 1.f : 0.f;
            p[e]   = raw * m;
            smask += raw * m;
            in_[e] = raw * (1.f - m);
            inSum += in_[e];
        }
        for (int e = 0; e < E_; ++e) inact[e] = in_[e] / (inSum + EPS_);

        // top-2 (first occurrence wins ties, matching lax.top_k)
        int i0 = 0; float v0 = p[0];
        for (int e = 1; e < E_; ++e) if (p[e] > v0) { v0 = p[e]; i0 = e; }
        int i1 = (i0 == 0) ? 1 : 0; float v1 = p[i1];
        for (int e = 0; e < E_; ++e) {
            if (e == i0 || e == i1) continue;
            if (p[e] > v1) { v1 = p[e]; i1 = e; }
        }
        const float den = v0 + v1 + EPS_;
        eIdx[2 * b + 0] = i0;  eIdx[2 * b + 1] = i1;
        pVal[2 * b + 0] = v0 / den;
        pVal[2 * b + 1] = v1 / den;

        atomicAdd(s_accum, smask);
    }
    __syncthreads();

    // selection embedding from inactive experts
    float acc = 0.f;
    for (int e = 0; e < E_; ++e)
        acc += selE[((size_t)b * E_ + e) * SE_ + t] * inact[e];
    sel_out[b * SE_ + t] = acc;
}

// =====================================================================
// Kernel 2: guide loss = (1 - s/B)^2
// =====================================================================
__global__ void moe_finalize(const float* __restrict__ s_accum,
                             float* __restrict__ guide_out) {
    const float s = s_accum[0] / (float)B_;
    const float g = 1.0f - s;
    guide_out[0] = g * g;
}

// =====================================================================
// Kernel 3: batched GEMM  out[b] = bf16( X[b] @ (p0 W[e0] + p1 W[e1]) + beff )
//   grid = (D/TN, L/TM, B), block = 256 (8 waves of 32)
//   wave (wm = w&3, wn = w>>2) owns a 32x64 slab -> 2x4 WMMA accumulators
// =====================================================================
__global__ __launch_bounds__(256)
void moe_gemm(const float* __restrict__ X,     // [B,L,F]
              const float* __restrict__ W,     // [E,F,D]
              const float* __restrict__ bias,  // [E,D]
              const int*   __restrict__ eIdx,  // [B,2]
              const float* __restrict__ pVal,  // [B,2]
              bf16_t* __restrict__ out)        // [B,L,D] bf16
{
    __shared__ bf16_t As[TM][LDS_STRIDE];   // A tile, row-major (k contiguous)
    __shared__ bf16_t Bs[TN][LDS_STRIDE];   // B tile, transposed (k contiguous per n)

    const int b    = blockIdx.z;
    const int m0   = blockIdx.y * TM;
    const int n0   = blockIdx.x * TN;
    const int tid  = threadIdx.x;
    const int lane = tid & 31;
    const int wave = tid >> 5;
    const int wm   = wave & 3;     // 0..3 : 32-row slab
    const int wn   = wave >> 2;    // 0..1 : 64-col slab
    const int half = lane >> 4;    // 0/1 : half-wave (K-group selector)
    const int l16  = lane & 15;

    const int   e0 = eIdx[2 * b + 0];
    const int   e1 = eIdx[2 * b + 1];
    const float p0 = pVal[2 * b + 0];
    const float p1 = pVal[2 * b + 1];

    const float* __restrict__ Xb = X + (size_t)b  * L_ * F_;
    const float* __restrict__ W0 = W + (size_t)e0 * F_ * D_;
    const float* __restrict__ W1 = W + (size_t)e1 * F_ * D_;

    v8f acc[2][4];
#pragma unroll
    for (int i = 0; i < 2; ++i)
#pragma unroll
        for (int j = 0; j < 4; ++j)
            acc[i][j] = (v8f){0.f, 0.f, 0.f, 0.f, 0.f, 0.f, 0.f, 0.f};

    const int kTiles = (F_ + TK - 1) / TK;   // 29 (ragged tail of 4)

    for (int kt = 0; kt < kTiles; ++kt) {
        const int k0 = kt * TK;

        // ---- stage A tile: 128x32 f32 -> bf16 in LDS ----
#pragma unroll
        for (int j = 0; j < 4; ++j) {
            const int f   = tid + 256 * j;          // float4 index 0..1023
            const int row = f >> 3;                 // 0..127
            const int kq  = (f & 7) << 2;           // 0,4,..,28
            const int kg  = k0 + kq;                // F_ % 4 == 0 -> whole-float4 guard
            float4 v = make_float4(0.f, 0.f, 0.f, 0.f);
            if (kg < F_)
                v = *(const float4*)(Xb + (size_t)(m0 + row) * F_ + kg);
            As[row][kq + 0] = (bf16_t)v.x;
            As[row][kq + 1] = (bf16_t)v.y;
            As[row][kq + 2] = (bf16_t)v.z;
            As[row][kq + 3] = (bf16_t)v.w;
        }

        // ---- stage B tile: blend two experts, 32x128 -> transposed bf16 ----
#pragma unroll
        for (int j = 0; j < 4; ++j) {
            const int f  = tid + 256 * j;           // float4 index 0..1023
            const int kr = f >> 5;                  // 0..31
            const int dq = (f & 31) << 2;           // 0,4,..,124
            const int kg = k0 + kr;
            float4 v = make_float4(0.f, 0.f, 0.f, 0.f);
            if (kg < F_) {
                const size_t off = (size_t)kg * D_ + n0 + dq;
                const float4 a = *(const float4*)(W0 + off);
                const float4 c = *(const float4*)(W1 + off);
                v.x = p0 * a.x + p1 * c.x;
                v.y = p0 * a.y + p1 * c.y;
                v.z = p0 * a.z + p1 * c.z;
                v.w = p0 * a.w + p1 * c.w;
            }
            Bs[dq + 0][kr] = (bf16_t)v.x;
            Bs[dq + 1][kr] = (bf16_t)v.y;
            Bs[dq + 2][kr] = (bf16_t)v.z;
            Bs[dq + 3][kr] = (bf16_t)v.w;
        }

        // prefetch hint for next K tile (emits global_prefetch_b8)
        if (k0 + TK < F_) {
            __builtin_prefetch(Xb + (size_t)(m0 + (tid >> 1)) * F_ + k0 + TK, 0, 1);
            __builtin_prefetch(W0 + (size_t)(k0 + TK + (tid & 31)) * D_ + n0, 0, 1);
        }

        __syncthreads();

        // ---- build WMMA fragments (ISA 7.12.2 16-bit layouts, wave32) ----
        v16bf aF[2];
        v16bf bF[4];
#pragma unroll
        for (int i = 0; i < 2; ++i) {
            const int mr = wm * 32 + i * 16 + l16;
#pragma unroll
            for (int v = 0; v < 8; ++v) {
                // A 16x32: VGPR v<4 -> K=2v(+1), v>=4 -> K=8+2v(+1); +8 for lanes 16-31
                const int kk = 2 * v + ((v >= 4) ? 8 : 0) + 8 * half;
                aF[i][2 * v + 0] = As[mr][kk + 0];
                aF[i][2 * v + 1] = As[mr][kk + 1];
            }
        }
#pragma unroll
        for (int jn = 0; jn < 4; ++jn) {
            const int nr = wn * 64 + jn * 16 + l16;
#pragma unroll
            for (int v = 0; v < 8; ++v) {
                // B 32x16: VGPR v<4 -> K=2v(+1), v>=4 -> K=8+2v(+1); +8 for lanes 16-31
                const int kk = 2 * v + ((v >= 4) ? 8 : 0) + 8 * half;
                bF[jn][2 * v + 0] = Bs[nr][kk + 0];
                bF[jn][2 * v + 1] = Bs[nr][kk + 1];
            }
        }

        // ---- 8 matrix ops per wave per K-step ----
#pragma unroll
        for (int i = 0; i < 2; ++i)
#pragma unroll
            for (int jn = 0; jn < 4; ++jn)
                acc[i][jn] = __builtin_amdgcn_wmma_f32_16x16x32_bf16(
                    false, aF[i], false, bF[jn],
                    (short)0, acc[i][jn], false, false);

        __syncthreads();
    }

    // ---- epilogue: add blended bias, convert to bf16, store ----
#pragma unroll
    for (int jn = 0; jn < 4; ++jn) {
        const int nCol = n0 + wn * 64 + jn * 16 + l16;
        const float be = p0 * bias[e0 * D_ + nCol] + p1 * bias[e1 * D_ + nCol];
#pragma unroll
        for (int i = 0; i < 2; ++i) {
            const int mBase = m0 + wm * 32 + i * 16 + 8 * half;  // C/D: M = r + 8*half
#pragma unroll
            for (int r = 0; r < 8; ++r) {
                out[(size_t)b * L_ * D_ + (size_t)(mBase + r) * D_ + nCol] =
                    (bf16_t)(acc[i][jn][r] + be);
            }
        }
    }
}

// =====================================================================
// Host launcher
// =====================================================================
extern "C" void kernel_launch(void* const* d_in, const int* in_sizes, int n_in,
                              void* d_out, int out_size, void* d_ws, size_t ws_size,
                              hipStream_t stream) {
    const float* X      = (const float*)d_in[0];  // cycle_curve_data [B,L,F]
    const float* logits = (const float*)d_in[1];  // [B,E]
    const int*   masks  = (const int*)  d_in[2];  // [B,E]
    const float* selE   = (const float*)d_in[3];  // [B,E,SE]
    const float* W      = (const float*)d_in[4];  // [E,F,D]
    const float* bias   = (const float*)d_in[5];  // [E,D]

    // workspace layout
    float* wsf     = (float*)d_ws;
    float* s_accum = wsf;                              // [1]
    int*   eIdx    = (int*)(wsf + 16);                 // [B,2]
    float* pVal    = (float*)(wsf + 16 + 2 * B_);      // [B,2]

    // output layout (tuple, concatenated): bf16 final_out, f32 guide, f32 sel_emb
    bf16_t* out_main  = (bf16_t*)d_out;
    float*  guide_out = (float*)((char*)d_out + (size_t)B_ * L_ * D_ * sizeof(bf16_t));
    float*  sel_out   = guide_out + 1;

    hipLaunchKernelGGL(moe_zero, dim3(1), dim3(1), 0, stream, s_accum);
    hipLaunchKernelGGL(moe_setup, dim3(B_), dim3(SE_), 0, stream,
                       logits, masks, selE, s_accum, eIdx, pVal, sel_out);
    hipLaunchKernelGGL(moe_finalize, dim3(1), dim3(1), 0, stream, s_accum, guide_out);

    dim3 grid(D_ / TN, L_ / TM, B_);   // (6, 8, 64)
    hipLaunchKernelGGL(moe_gemm, grid, dim3(256), 0, stream,
                       X, W, bias, eIdx, pVal, out_main);
}
